// PRISMMultiTaskNN_69758858821908
// MI455X (gfx1250) — matrix operand order repaired and checked by reading, not compiled, not run on hardware
//
#include <hip/hip_runtime.h>
#include <cstdint>

// ---------------- problem constants ----------------
#define BB     8192
#define D_IN   19200
#define D1     512
#define D2     256
#define D_PW   128
#define N_PW   25
#define N_DRUG 1448

// ---------------- WMMA vector types ----------------
typedef __attribute__((ext_vector_type(16))) __bf16 bf16x16;
typedef __attribute__((ext_vector_type(2)))  __bf16 bf16x2;
typedef __attribute__((ext_vector_type(8)))  float  floatx8;

union FragBF {
    uint4   q[2];   // two 16B halves (b128 loads)
    bf16x16 v;      // operand for v_wmma_f32_16x16x32_bf16
};
static_assert(sizeof(FragBF) == 32, "frag size");

// fp32 -> bf16 (RNE) helpers; pack_bf2 maps to a single v_cvt_pk_bf16_f32
__device__ __forceinline__ unsigned short bfbits(float f) {
    __bf16 h = (__bf16)f;
    return __builtin_bit_cast(unsigned short, h);
}
__device__ __forceinline__ unsigned int pack_bf2(float lo, float hi) {
    bf16x2 p;
    p.x = (__bf16)lo;
    p.y = (__bf16)hi;
    return __builtin_bit_cast(unsigned int, p);
}

// LDS tile geometry for GEMM1 A: K-step 32 bf16 + 8 pad => 40 u16 (80B stride, 16B aligned)
#define TK      32
#define KSTRIDE 40

// ============================================================================
// Prep: convert W[K,N] fp32 -> bf16, tiled as Wt[(k/32)*N + n][k%32] so a wave's
// WMMA B-fragment (16 contiguous K for column n) is two contiguous b128 loads.
// Runs once; W1t (19.7MB) and W2t (0.26MB) then live in the 192MB L2.
// ============================================================================
__global__ __launch_bounds__(256) void pack_wt_kernel(
    const float* __restrict__ W, unsigned short* __restrict__ Wt, int K, int N)
{
    const int ng  = N >> 2;
    const int idx = blockIdx.x * 256 + threadIdx.x;    // one per (k, n/4)
    if (idx >= K * ng) return;
    const int k  = idx / ng;
    const int n4 = (idx % ng) * 4;
    const float4 v = *(const float4*)(W + (size_t)k * N + n4);
    unsigned short* dst = Wt + ((size_t)(k >> 5) * N) * 32 + (k & 31);
    dst[(size_t)(n4 + 0) * 32] = bfbits(v.x);
    dst[(size_t)(n4 + 1) * 32] = bfbits(v.y);
    dst[(size_t)(n4 + 2) * 32] = bfbits(v.z);
    dst[(size_t)(n4 + 3) * 32] = bfbits(v.w);
}

// ============================================================================
// GEMM1: H1[8192,512](bf16) = relu( X[8192,19200](f32) @ W1 + b1 )
// 128x128 tile per 256-thread block; A (x, converted to bf16) through
// double-buffered LDS (reused by 4 waves); B fragments streamed straight from
// the L2-resident pre-tiled W1t. Main loop manually unrolled by 2 so the two
// B-fragment register sets alternate with no copy movs.
// ============================================================================
__global__ __launch_bounds__(256) void gemm1_kernel(
    const float* __restrict__ X, const unsigned short* __restrict__ W1t,
    const float* __restrict__ b1, unsigned short* __restrict__ H1)
{
    __shared__ __align__(16) unsigned short As[2][128 * KSTRIDE]; // [m][k] bf16

    const int tid  = threadIdx.x;
    const int lane = tid & 31;
    const int wave = tid >> 5;
    const int wm   = wave >> 2;   // 0..1 : 64-row slab
    const int wn   = wave & 3;    // 0..3 : 32-col slab
    const int m0   = blockIdx.y * 128;
    const int n0   = blockIdx.x * 128;

    const int aRow = tid >> 1;           // 0..127
    const int aK   = (tid & 1) * 16;     // 0 or 16

    const int hi = lane >> 4;            // half-wave select (ISA 16-bit layout)
    const int lr = lane & 15;

    floatx8 acc[4][2];
#pragma unroll
    for (int m = 0; m < 4; ++m)
#pragma unroll
        for (int n = 0; n < 2; ++n)
            acc[m][n] = (floatx8){0.f,0.f,0.f,0.f,0.f,0.f,0.f,0.f};

    float4 pa[4];   // A prefetch: 16 fp32

    auto loadAglb = [&](int kk) {
        const float* xp = X + (size_t)(m0 + aRow) * D_IN + kk * TK + aK;
        pa[0] = ((const float4*)xp)[0];
        pa[1] = ((const float4*)xp)[1];
        pa[2] = ((const float4*)xp)[2];
        pa[3] = ((const float4*)xp)[3];
        __builtin_prefetch(xp + 2 * TK, 0, 1);   // global_prefetch_b8
    };
    auto storeA = [&](int buf) {
        uint4 q0 = make_uint4(pack_bf2(pa[0].x, pa[0].y), pack_bf2(pa[0].z, pa[0].w),
                              pack_bf2(pa[1].x, pa[1].y), pack_bf2(pa[1].z, pa[1].w));
        uint4 q1 = make_uint4(pack_bf2(pa[2].x, pa[2].y), pack_bf2(pa[2].z, pa[2].w),
                              pack_bf2(pa[3].x, pa[3].y), pack_bf2(pa[3].z, pa[3].w));
        unsigned short* ap = &As[buf][aRow * KSTRIDE + aK];
        ((uint4*)ap)[0] = q0;
        ((uint4*)ap)[1] = q1;
    };

    // B fragment base: W1t element ((kk*512 + n)*32 + ki); per-lane column n,
    // fragment element e -> K = e + 16*hi (two contiguous b128 halves).
    const unsigned short* bBase =
        W1t + ((size_t)(n0 + wn * 32 + lr)) * 32 + hi * 16;

    auto loadB = [&](FragBF* bf, int kk) {
        const unsigned short* p0 = bBase + (size_t)kk * (D1 * 32);
        bf[0].q[0] = *(const uint4*)p0;
        bf[0].q[1] = *(const uint4*)(p0 + 8);
        const unsigned short* p1 = p0 + 16 * 32;   // +16 columns
        bf[1].q[0] = *(const uint4*)p1;
        bf[1].q[1] = *(const uint4*)(p1 + 8);
    };

    auto compute = [&](int buf, FragBF* bf) {
        FragBF af[4];
#pragma unroll
        for (int m = 0; m < 4; ++m) {
            const unsigned short* p = &As[buf][(wm * 64 + m * 16 + lr) * KSTRIDE];
            af[m].q[0] = *(const uint4*)(p + hi * 16);       // K 0..7 (+16 upper half)
            af[m].q[1] = *(const uint4*)(p + hi * 16 + 8);   // K 8..15 (+16 upper half)
        }
#pragma unroll
        for (int m = 0; m < 4; ++m)
#pragma unroll
            for (int n = 0; n < 2; ++n)
                acc[m][n] = __builtin_amdgcn_wmma_f32_16x16x32_bf16(
                    false, af[m].v, false, bf[n].v, (short)0, acc[m][n], false, false);
    };

    const int KSTEPS = D_IN / TK;  // 600 (even)
    FragBF b0[2], b1f[2];
    loadAglb(0);
    loadB(b0, 0);
    storeA(0);
    __syncthreads();
    for (int kk = 0; kk < KSTEPS; kk += 2) {
        // ---- even step: compute LDS buf0 with b0, prefetch kk+1 ----
        loadAglb(kk + 1);
        loadB(b1f, kk + 1);
        compute(0, b0);
        storeA(1);
        __syncthreads();
        // ---- odd step: compute LDS buf1 with b1f, prefetch kk+2 ----
        const bool more = (kk + 2) < KSTEPS;
        if (more) { loadAglb(kk + 2); loadB(b0, kk + 2); }
        compute(1, b1f);
        if (more) storeA(0);
        __syncthreads();
    }

    // epilogue: bias + relu, store bf16 (C/D layout: M = j + 8*hi, N = lr)
#pragma unroll
    for (int n = 0; n < 2; ++n) {
        const int   N    = n0 + wn * 32 + n * 16 + lr;
        const float bias = b1[N];
#pragma unroll
        for (int m = 0; m < 4; ++m) {
            const int Mb = m0 + wm * 64 + m * 16 + hi * 8;
#pragma unroll
            for (int j = 0; j < 8; ++j) {
                float v = acc[m][n][j] + bias;
                v = v > 0.f ? v : 0.f;
                H1[(size_t)(Mb + j) * D1 + N] = bfbits(v);
            }
        }
    }
}

// ============================================================================
// GEMM2: H2[8192,256](f32) = relu( H1[8192,512](bf16) @ W2 + b2 )
// Both operands already bf16 in WMMA-fragment-friendly layouts (H1 row-major,
// W2t tiled) -> pure register-pipelined global b128 loads + WMMA. No LDS.
// ============================================================================
__global__ __launch_bounds__(256) void gemm2_kernel(
    const unsigned short* __restrict__ H1, const unsigned short* __restrict__ W2t,
    const float* __restrict__ b2, float* __restrict__ H2)
{
    const int tid  = threadIdx.x;
    const int lane = tid & 31;
    const int wave = tid >> 5;
    const int wm   = wave >> 2;
    const int wn   = wave & 3;
    const int m0   = blockIdx.y * 128;
    const int n0   = blockIdx.x * 128;

    const int hi = lane >> 4;
    const int lr = lane & 15;

    floatx8 acc[4][2];
#pragma unroll
    for (int m = 0; m < 4; ++m)
#pragma unroll
        for (int n = 0; n < 2; ++n)
            acc[m][n] = (floatx8){0.f,0.f,0.f,0.f,0.f,0.f,0.f,0.f};

    const unsigned short* aBase[4];
#pragma unroll
    for (int m = 0; m < 4; ++m)
        aBase[m] = H1 + (size_t)(m0 + wm * 64 + m * 16 + lr) * D1 + hi * 16;
    const unsigned short* bBase =
        W2t + ((size_t)(n0 + wn * 32 + lr)) * 32 + hi * 16;

    auto loadAB = [&](FragBF* af, FragBF* bf, int kk) {
#pragma unroll
        for (int m = 0; m < 4; ++m) {
            const unsigned short* p = aBase[m] + kk * TK;
            af[m].q[0] = *(const uint4*)p;           // K 0..7  (+16 upper half)
            af[m].q[1] = *(const uint4*)(p + 8);     // K 8..15 (+16 upper half)
        }
#pragma unroll
        for (int n = 0; n < 2; ++n) {
            const unsigned short* p = bBase + (size_t)kk * (D2 * 32) + (size_t)n * (16 * 32);
            bf[n].q[0] = *(const uint4*)p;
            bf[n].q[1] = *(const uint4*)(p + 8);
        }
    };
    auto domma = [&](FragBF* af, FragBF* bf) {
#pragma unroll
        for (int m = 0; m < 4; ++m)
#pragma unroll
            for (int n = 0; n < 2; ++n)
                acc[m][n] = __builtin_amdgcn_wmma_f32_16x16x32_bf16(
                    false, af[m].v, false, bf[n].v, (short)0, acc[m][n], false, false);
    };

    const int KSTEPS = D1 / TK;  // 16
    FragBF a0[4], b0[2], a1[4], b1f[2];
    loadAB(a0, b0, 0);
    for (int kk = 0; kk < KSTEPS; kk += 2) {
        if (kk + 1 < KSTEPS) loadAB(a1, b1f, kk + 1);
        domma(a0, b0);
        if (kk + 2 < KSTEPS) loadAB(a0, b0, kk + 2);
        if (kk + 1 < KSTEPS) domma(a1, b1f);
    }

#pragma unroll
    for (int n = 0; n < 2; ++n) {
        const int   N    = n0 + wn * 32 + n * 16 + lr;
        const float bias = b2[N];
#pragma unroll
        for (int m = 0; m < 4; ++m) {
            const int Mb = m0 + wm * 64 + m * 16 + hi * 8;
#pragma unroll
            for (int j = 0; j < 8; ++j) {
                float v = acc[m][n][j] + bias;
                H2[(size_t)(Mb + j) * D2 + N] = v > 0.f ? v : 0.f;
            }
        }
    }
}

// ============================================================================
// Heads: per-sample pathway gather + per-drug dot.
// out[b] = sum_j relu( Wp[p][j,:] . h2[b] + bp[p][j] ) * Wd[d][j] + bd[d]
// One wave32 per sample; Wp (3.3MB) is L2-resident.
// ============================================================================
__global__ __launch_bounds__(256) void head_kernel(
    const float* __restrict__ H2, const float* __restrict__ Wp,
    const float* __restrict__ bp, const float* __restrict__ Wd,
    const float* __restrict__ bd, const long long* __restrict__ didx,
    const int* __restrict__ dpw, float* __restrict__ out)
{
    __shared__ __align__(16) float hsh[8][D2];
    const int wv   = threadIdx.x >> 5;
    const int lane = threadIdx.x & 31;
    const int b    = blockIdx.x * 8 + wv;

    {
        const float4* src = (const float4*)(H2 + (size_t)b * D2 + lane * 8);
        float4*       dst = (float4*)&hsh[wv][lane * 8];
        dst[0] = src[0];
        dst[1] = src[1];
    }
    __syncthreads();

    const long long d = didx[b];
    const int       p = dpw[d];
    const float* wpbase = Wp + (size_t)p * (D_PW * D2);

    float total = 0.f;
#pragma unroll
    for (int jj = 0; jj < 4; ++jj) {
        const int j = jj * 32 + lane;
        const float4* wrow = (const float4*)(wpbase + (size_t)j * D2);
        const float4* hq   = (const float4*)&hsh[wv][0];
        float acc = 0.f;
#pragma unroll 8
        for (int k4 = 0; k4 < D2 / 4; ++k4) {
            const float4 w = wrow[k4];
            const float4 h = hq[k4];
            acc += w.x * h.x + w.y * h.y + w.z * h.z + w.w * h.w;
        }
        float r = acc + bp[p * D_PW + j];
        r = r > 0.f ? r : 0.f;
        total += r * Wd[(size_t)d * D_PW + j];
    }
#pragma unroll
    for (int off = 16; off > 0; off >>= 1)
        total += __shfl_down(total, off, 32);
    if (lane == 0) out[b] = total + bd[d];
}

// ============================================================================
extern "C" void kernel_launch(void* const* d_in, const int* in_sizes, int n_in,
                              void* d_out, int out_size, void* d_ws, size_t ws_size,
                              hipStream_t stream) {
    const float*     x    = (const float*)d_in[0];
    const float*     W1   = (const float*)d_in[1];
    const float*     b1   = (const float*)d_in[2];
    const float*     W2   = (const float*)d_in[3];
    const float*     b2   = (const float*)d_in[4];
    const float*     Wp   = (const float*)d_in[5];
    const float*     bp   = (const float*)d_in[6];
    const float*     Wd   = (const float*)d_in[7];
    const float*     bd   = (const float*)d_in[8];
    const long long* didx = (const long long*)d_in[9];   // int64 per reference
    const int*       dpw  = (const int*)d_in[10];
    float*           out  = (float*)d_out;

    // workspace layout (bytes):
    //   H1  bf16 [8192,512]    @ 0            (8,388,608)
    //   H2  f32  [8192,256]    @ 8,388,608    (8,388,608)
    //   W1t bf16 tiled         @ 16,777,216   (19,660,800)
    //   W2t bf16 tiled         @ 36,438,016   (   262,144)
    char* ws = (char*)d_ws;
    unsigned short* H1  = (unsigned short*)(ws);
    float*          H2  = (float*)(ws + (size_t)BB * D1 * 2);
    unsigned short* W1t = (unsigned short*)(ws + (size_t)BB * D1 * 2 + (size_t)BB * D2 * 4);
    unsigned short* W2t = (unsigned short*)(ws + (size_t)BB * D1 * 2 + (size_t)BB * D2 * 4
                                               + (size_t)D_IN * D1 * 2);

    // one-time weight convert+tile (tiny; W1t/W2t then served from L2)
    pack_wt_kernel<<<(D_IN * (D1 / 4) + 255) / 256, 256, 0, stream>>>(W1, W1t, D_IN, D1);
    pack_wt_kernel<<<(D1 * (D2 / 4) + 255) / 256, 256, 0, stream>>>(W2, W2t, D1, D2);

    dim3 g1(D1 / 128, BB / 128);   // 4 x 64
    gemm1_kernel<<<g1, 256, 0, stream>>>(x, W1t, b1, H1);

    dim3 g2(D2 / 128, BB / 128);   // 2 x 64
    gemm2_kernel<<<g2, 256, 0, stream>>>(H1, W2t, b2, H2);

    head_kernel<<<BB / 8, 256, 0, stream>>>(H2, Wp, bp, Wd, bd, didx, dpw, out);
}
